// QuantumNATHybrid_65481071405998
// MI455X (gfx1250) — matrix operand order repaired and checked by reading, not compiled, not run on hardware
//
#include <hip/hip_runtime.h>
#include <hip/hip_bf16.h>

typedef __attribute__((ext_vector_type(16))) _Float16 v16h;
typedef __attribute__((ext_vector_type(8)))  _Float16 v8h;
typedef __attribute__((ext_vector_type(8)))  float    v8f;

#define BATCH 8192
#define EPSV 1e-5f

// ---------------------------------------------------------------------------
// Kernel 1: conv1 (1->8, 3x3 SAME) batch statistics (sum, sumsq per channel)
// one thread per (b, y, x); grid covers B*784 exactly.
// ---------------------------------------------------------------------------
__global__ __launch_bounds__(256)
void k1_conv1_stats(const float* __restrict__ x, const float* __restrict__ w1,
                    float* __restrict__ stats /* [0..7]=sum [8..15]=sumsq */) {
    __shared__ float w1s[72];
    __shared__ float acc[16];
    for (int i = threadIdx.x; i < 72; i += 256) w1s[i] = w1[i];
    if (threadIdx.x < 16) acc[threadIdx.x] = 0.f;
    __syncthreads();

    int tid = blockIdx.x * 256 + threadIdx.x;
    int b = tid / 784;
    int p = tid % 784;
    int y = p / 28, xx = p % 28;
    const float* img = x + (size_t)b * 784;

    float v[3][3];
    for (int ky = 0; ky < 3; ky++)
        for (int kx = 0; kx < 3; kx++) {
            int yy = y + ky - 1, xc = xx + kx - 1;
            v[ky][kx] = (yy >= 0 && yy < 28 && xc >= 0 && xc < 28) ? img[yy * 28 + xc] : 0.f;
        }

    float ls[8], lq[8];
    for (int c = 0; c < 8; c++) {
        float a = 0.f;
        for (int ky = 0; ky < 3; ky++)
            for (int kx = 0; kx < 3; kx++)
                a = fmaf(w1s[c * 9 + ky * 3 + kx], v[ky][kx], a);
        ls[c] = a; lq[c] = a * a;
    }
    for (int c = 0; c < 8; c++) {
        atomicAdd(&acc[c], ls[c]);
        atomicAdd(&acc[8 + c], lq[c]);
    }
    __syncthreads();
    if (threadIdx.x < 16) atomicAdd(&stats[threadIdx.x], acc[threadIdx.x]);
}

// ---------------------------------------------------------------------------
// Kernel 2: conv1 + BN1 + ReLU + 2x2 maxpool -> a1 (B,8,14,14)
// one thread per (b, py, px); grid covers B*196 exactly.
// ---------------------------------------------------------------------------
__global__ __launch_bounds__(256)
void k2_conv1_bn_pool(const float* __restrict__ x, const float* __restrict__ w1,
                      const float* __restrict__ g, const float* __restrict__ bb,
                      const float* __restrict__ stats, float* __restrict__ a1) {
    __shared__ float w1s[72];
    __shared__ float sc[8], sh[8];
    for (int i = threadIdx.x; i < 72; i += 256) w1s[i] = w1[i];
    if (threadIdx.x < 8) {
        const float invN = 1.f / (float)(BATCH * 784);
        float m = stats[threadIdx.x] * invN;
        float vv = stats[8 + threadIdx.x] * invN - m * m;
        float s = g[threadIdx.x] * rsqrtf(vv + EPSV);
        sc[threadIdx.x] = s;
        sh[threadIdx.x] = bb[threadIdx.x] - m * s;
    }
    __syncthreads();

    int tid = blockIdx.x * 256 + threadIdx.x;
    int b = tid / 196;
    int p = tid % 196;
    int py = p / 14, px = p % 14;
    const float* img = x + (size_t)b * 784;

    float v[4][4];
    int r0 = 2 * py - 1, c0 = 2 * px - 1;
    for (int i = 0; i < 4; i++)
        for (int j = 0; j < 4; j++) {
            int yy = r0 + i, xc = c0 + j;
            v[i][j] = (yy >= 0 && yy < 28 && xc >= 0 && xc < 28) ? img[yy * 28 + xc] : 0.f;
        }

    for (int c = 0; c < 8; c++) {
        float m = -1e30f;
        for (int dy = 0; dy < 2; dy++)
            for (int dx = 0; dx < 2; dx++) {
                float a = 0.f;
                for (int ky = 0; ky < 3; ky++)
                    for (int kx = 0; kx < 3; kx++)
                        a = fmaf(w1s[c * 9 + ky * 3 + kx], v[dy + ky][dx + kx], a);
                m = fmaxf(m, fmaf(a, sc[c], sh[c]));
            }
        a1[(((size_t)b * 8 + c) * 14 + py) * 14 + px] = fmaxf(m, 0.f);
    }
}

// ---------------------------------------------------------------------------
// Kernel 3: conv2 (8->16, 3x3 SAME) batch statistics
// one thread per (b, y, x) on the 14x14 grid; B*196 exact.
// ---------------------------------------------------------------------------
__global__ __launch_bounds__(256)
void k3_conv2_stats(const float* __restrict__ a1, const float* __restrict__ w2,
                    float* __restrict__ stats /* [16..31]=sum [32..47]=sumsq */) {
    __shared__ float w2s[1152];
    __shared__ float acc[32];
    for (int i = threadIdx.x; i < 1152; i += 256) w2s[i] = w2[i];
    if (threadIdx.x < 32) acc[threadIdx.x] = 0.f;
    __syncthreads();

    int tid = blockIdx.x * 256 + threadIdx.x;
    int b = tid / 196;
    int p = tid % 196;
    int y = p / 14, xx = p % 14;

    float out[16];
    for (int oc = 0; oc < 16; oc++) out[oc] = 0.f;

    for (int ic = 0; ic < 8; ic++) {
        const float* ch = a1 + ((size_t)b * 8 + ic) * 196;
        float v[3][3];
        for (int ky = 0; ky < 3; ky++)
            for (int kx = 0; kx < 3; kx++) {
                int yy = y + ky - 1, xc = xx + kx - 1;
                v[ky][kx] = (yy >= 0 && yy < 14 && xc >= 0 && xc < 14) ? ch[yy * 14 + xc] : 0.f;
            }
        for (int oc = 0; oc < 16; oc++) {
            const float* wk = &w2s[(oc * 8 + ic) * 9];
            float a = out[oc];
            for (int ky = 0; ky < 3; ky++)
                for (int kx = 0; kx < 3; kx++)
                    a = fmaf(wk[ky * 3 + kx], v[ky][kx], a);
            out[oc] = a;
        }
    }
    for (int oc = 0; oc < 16; oc++) {
        atomicAdd(&acc[oc], out[oc]);
        atomicAdd(&acc[16 + oc], out[oc] * out[oc]);
    }
    __syncthreads();
    if (threadIdx.x < 32) atomicAdd(&stats[16 + threadIdx.x], acc[threadIdx.x]);
}

// ---------------------------------------------------------------------------
// Kernel 4: conv2 + BN2 + ReLU + maxpool -> a2 flattened (B,784) NCHW order
// one thread per (b, py, px) on 7x7; B*49 exact.
// ---------------------------------------------------------------------------
__global__ __launch_bounds__(256)
void k4_conv2_bn_pool(const float* __restrict__ a1, const float* __restrict__ w2,
                      const float* __restrict__ g, const float* __restrict__ bb,
                      const float* __restrict__ stats, float* __restrict__ a2) {
    __shared__ float w2s[1152];
    __shared__ float sc[16], sh[16];
    for (int i = threadIdx.x; i < 1152; i += 256) w2s[i] = w2[i];
    if (threadIdx.x < 16) {
        const float invN = 1.f / (float)(BATCH * 196);
        float m = stats[16 + threadIdx.x] * invN;
        float vv = stats[32 + threadIdx.x] * invN - m * m;
        float s = g[threadIdx.x] * rsqrtf(vv + EPSV);
        sc[threadIdx.x] = s;
        sh[threadIdx.x] = bb[threadIdx.x] - m * s;
    }
    __syncthreads();

    int tid = blockIdx.x * 256 + threadIdx.x;
    int b = tid / 49;
    int p = tid % 49;
    int py = p / 7, px = p % 7;

    float acc[16][4];
    for (int oc = 0; oc < 16; oc++)
        for (int q = 0; q < 4; q++) acc[oc][q] = 0.f;

    int r0 = 2 * py - 1, c0 = 2 * px - 1;
    for (int ic = 0; ic < 8; ic++) {
        const float* ch = a1 + ((size_t)b * 8 + ic) * 196;
        float v[4][4];
        for (int i = 0; i < 4; i++)
            for (int j = 0; j < 4; j++) {
                int yy = r0 + i, xc = c0 + j;
                v[i][j] = (yy >= 0 && yy < 14 && xc >= 0 && xc < 14) ? ch[yy * 14 + xc] : 0.f;
            }
        for (int oc = 0; oc < 16; oc++) {
            const float* wk = &w2s[(oc * 8 + ic) * 9];
            for (int dy = 0; dy < 2; dy++)
                for (int dx = 0; dx < 2; dx++) {
                    float a = acc[oc][dy * 2 + dx];
                    for (int ky = 0; ky < 3; ky++)
                        for (int kx = 0; kx < 3; kx++)
                            a = fmaf(wk[ky * 3 + kx], v[dy + ky][dx + kx], a);
                    acc[oc][dy * 2 + dx] = a;
                }
        }
    }
    for (int oc = 0; oc < 16; oc++) {
        float m = -1e30f;
        for (int q = 0; q < 4; q++) m = fmaxf(m, fmaf(acc[oc][q], sc[oc], sh[oc]));
        // flattened (B,784): index = oc*49 + py*7 + px
        a2[(size_t)b * 784 + oc * 49 + py * 7 + px] = fmaxf(m, 0.f);
    }
}

// ---------------------------------------------------------------------------
// Kernel 5: map layer via WMMA:  xmap(B,4) = a2(B,784) @ map_w.T(784,4) + map_b
// One wave = one 16(batch) x 16(N padded) tile, K=784 = 24 full k-steps of 32
// plus one zero-padded tail step. B (4x784) is staged zero-padded in LDS as
// f16 (16 x 800), so the hot loop is pure b128 loads + cvt + wmma, no guards.
// Fragment packing per CDNA5 ISA 7.12.2 (16-bit A 16x32, B 32x16, f32 C/D).
// ---------------------------------------------------------------------------
#define BS_STRIDE 808   // 800 used + 8 pad halves (16B-multiple row stride)

__global__ __launch_bounds__(128)
void k5_map_wmma(const float* __restrict__ a2, const float* __restrict__ map_w,
                 const float* __restrict__ map_b, float* __restrict__ xmap) {
    __shared__ _Float16 Bs[16][BS_STRIDE];

    // Stage zero-padded B tile: rows 0..3 = map_w rows, rows 4..15 = 0,
    // K 784..799 = 0.  16*800 elements, 128 threads.
    for (int idx = threadIdx.x; idx < 16 * 800; idx += 128) {
        int nn = idx / 800, kk = idx % 800;
        float val = (nn < 4 && kk < 784) ? map_w[nn * 784 + kk] : 0.f;
        Bs[nn][kk] = (_Float16)val;
    }
    __syncthreads();

    int wave = threadIdx.x >> 5;
    int lane = threadIdx.x & 31;
    int tile = blockIdx.x * 4 + wave;      // 128 blocks * 4 waves * 16 rows = 8192
    int rowbase = tile * 16;
    int n = lane & 15;                     // A: M row; B/D: N column
    int hi = (lane >= 16) ? 1 : 0;

    const float4* arow4 = (const float4*)(a2 + (size_t)(rowbase + n) * 784);

    v8f c = {};

    // ---- 24 unguarded full K-steps (K 0..767) ----
    for (int k0 = 0; k0 < 768; k0 += 32) {
        int abase = k0 + hi * 8;           // multiple of 8 floats -> float4 aligned
        __builtin_prefetch((const float*)arow4 + abase + 256, 0, 3);

        float4 f0 = arow4[(abase >> 2)];
        float4 f1 = arow4[(abase >> 2) + 1];
        float4 f2 = arow4[(abase >> 2) + 4];   // abase + 16
        float4 f3 = arow4[(abase >> 2) + 5];

        v16h A;
        A[0] = (_Float16)f0.x;  A[1] = (_Float16)f0.y;
        A[2] = (_Float16)f0.z;  A[3] = (_Float16)f0.w;
        A[4] = (_Float16)f1.x;  A[5] = (_Float16)f1.y;
        A[6] = (_Float16)f1.z;  A[7] = (_Float16)f1.w;
        A[8] = (_Float16)f2.x;  A[9] = (_Float16)f2.y;
        A[10] = (_Float16)f2.z; A[11] = (_Float16)f2.w;
        A[12] = (_Float16)f3.x; A[13] = (_Float16)f3.y;
        A[14] = (_Float16)f3.z; A[15] = (_Float16)f3.w;

        // B fragment: N = lane%16, K = k0 + 8*hi + h (h<8), k0 + 16 + 8*hi + (h-8) (h>=8).
        // 16B-aligned LDS reads -> two ds_load_b128.
        const _Float16* bp = &Bs[n][k0 + hi * 8];
        v8h b0 = *(const v8h*)bp;
        v8h b1 = *(const v8h*)(bp + 16);
        v16h Bf;
        for (int h = 0; h < 8; h++) { Bf[h] = b0[h]; Bf[h + 8] = b1[h]; }

        c = __builtin_amdgcn_wmma_f32_16x16x32_f16(false, A, false, Bf,
                                                   (short)0, c, false, false);
    }

    // ---- tail K-step (K 768..783 valid, 784..799 zero) ----
    {
        int abase = 768 + hi * 8;          // 768..775 / 776..783: fully in range
        float4 f0 = arow4[(abase >> 2)];
        float4 f1 = arow4[(abase >> 2) + 1];

        v16h A;
        A[0] = (_Float16)f0.x;  A[1] = (_Float16)f0.y;
        A[2] = (_Float16)f0.z;  A[3] = (_Float16)f0.w;
        A[4] = (_Float16)f1.x;  A[5] = (_Float16)f1.y;
        A[6] = (_Float16)f1.z;  A[7] = (_Float16)f1.w;
        for (int h = 8; h < 16; h++) A[h] = (_Float16)0.f;  // K >= 784 for both halves

        const _Float16* bp = &Bs[n][768 + hi * 8];          // zero-padded in LDS
        v8h b0 = *(const v8h*)bp;
        v8h b1 = *(const v8h*)(bp + 16);
        v16h Bf;
        for (int h = 0; h < 8; h++) { Bf[h] = b0[h]; Bf[h + 8] = b1[h]; }

        c = __builtin_amdgcn_wmma_f32_16x16x32_f16(false, A, false, Bf,
                                                   (short)0, c, false, false);
    }

    // C/D layout: VGPR r holds M=r (lanes 0-15) / M=r+8 (lanes 16-31); N = lane%16
    if (n < 4) {
        float bias = map_b[n];
        for (int r = 0; r < 8; r++) {
            int m = r + (hi ? 8 : 0);
            xmap[(size_t)(rowbase + m) * 4 + n] = c[r] + bias;
        }
    }
}

// ---------------------------------------------------------------------------
// Kernel 6: exact 4-qubit statevector simulation + bnq batch statistics
// one thread per batch element (8192 / 256 = 32 blocks).
// ---------------------------------------------------------------------------
__global__ __launch_bounds__(256)
void k6_quantum(const float* __restrict__ xmap, const float* __restrict__ qw,
                float* __restrict__ qout, float* __restrict__ stats /*48..55*/) {
    __shared__ float acc[8];
    if (threadIdx.x < 8) acc[threadIdx.x] = 0.f;
    __syncthreads();

    int b = blockIdx.x * 256 + threadIdx.x;
    float xm[4];
    for (int i = 0; i < 4; i++) xm[i] = xmap[(size_t)b * 4 + i];

    float re[16], im[16];
    for (int i = 0; i < 16; i++) { re[i] = 0.f; im[i] = 0.f; }
    re[0] = 1.f;

    // RY(x_i) on wire i   (wire 0 = MSB of the 4-bit state index)
    for (int w = 0; w < 4; w++) {
        float half = 0.5f * xm[w];
        float cg = __cosf(half), sg = __sinf(half);
        int tb = 1 << (3 - w);
        for (int idx = 0; idx < 16; idx++) {
            if (idx & tb) continue;
            int j = idx | tb;
            float r0 = re[idx], i0 = im[idx], r1 = re[j], i1 = im[j];
            re[idx] = cg * r0 - sg * r1;  im[idx] = cg * i0 - sg * i1;
            re[j]   = sg * r0 + cg * r1;  im[j]   = sg * i0 + cg * i1;
        }
    }
    // RZ(w_i): phase exp(-i w/2) on |0>, exp(+i w/2) on |1>
    for (int w = 0; w < 4; w++) {
        float a = 0.5f * qw[w];
        float cp = __cosf(a), sp = __sinf(a);
        int tb = 1 << (3 - w);
        for (int idx = 0; idx < 16; idx++) {
            float pi = (idx & tb) ? sp : -sp;
            float r = re[idx], i2 = im[idx];
            re[idx] = r * cp - i2 * pi;
            im[idx] = r * pi + i2 * cp;
        }
    }
    // CNOT chain 0->1->2->3
    for (int k = 0; k < 3; k++) {
        int cb = 1 << (3 - k), tb = 1 << (3 - (k + 1));
        for (int idx = 0; idx < 16; idx++) {
            if ((idx & cb) && !(idx & tb)) {
                int j = idx | tb;
                float tr = re[idx], ti = im[idx];
                re[idx] = re[j]; im[idx] = im[j];
                re[j] = tr; im[j] = ti;
            }
        }
    }
    // <Z_i>
    float z[4] = {0.f, 0.f, 0.f, 0.f};
    for (int idx = 0; idx < 16; idx++) {
        float p = re[idx] * re[idx] + im[idx] * im[idx];
        for (int w = 0; w < 4; w++) z[w] += (idx & (1 << (3 - w))) ? -p : p;
    }
    for (int w = 0; w < 4; w++) {
        qout[(size_t)b * 4 + w] = z[w];
        atomicAdd(&acc[w], z[w]);
        atomicAdd(&acc[4 + w], z[w] * z[w]);
    }
    __syncthreads();
    if (threadIdx.x < 8) atomicAdd(&stats[48 + threadIdx.x], acc[threadIdx.x]);
}

// ---------------------------------------------------------------------------
// Kernel 7: bnq + fc1 (4->32) + ReLU + fc2 (32->4) -> logits
// ---------------------------------------------------------------------------
__global__ __launch_bounds__(256)
void k7_head(const float* __restrict__ qv, const float* __restrict__ stats,
             const float* __restrict__ gq, const float* __restrict__ bq,
             const float* __restrict__ fc1w, const float* __restrict__ fc1b,
             const float* __restrict__ fc2w, const float* __restrict__ fc2b,
             float* __restrict__ out) {
    __shared__ float w1s[128], w2s[128], b1s[32], b2s[4], sc[4], sh[4];
    for (int i = threadIdx.x; i < 128; i += 256) { w1s[i] = fc1w[i]; w2s[i] = fc2w[i]; }
    if (threadIdx.x < 32) b1s[threadIdx.x] = fc1b[threadIdx.x];
    if (threadIdx.x < 4) {
        b2s[threadIdx.x] = fc2b[threadIdx.x];
        const float invN = 1.f / (float)BATCH;
        float m = stats[48 + threadIdx.x] * invN;
        float vv = stats[52 + threadIdx.x] * invN - m * m;
        float s = gq[threadIdx.x] * rsqrtf(vv + EPSV);
        sc[threadIdx.x] = s;
        sh[threadIdx.x] = bq[threadIdx.x] - m * s;
    }
    __syncthreads();

    int b = blockIdx.x * 256 + threadIdx.x;
    float qn[4];
    for (int i = 0; i < 4; i++)
        qn[i] = fmaf(qv[(size_t)b * 4 + i], sc[i], sh[i]);

    float h[32];
    for (int j = 0; j < 32; j++) {
        float a = b1s[j];
        for (int k = 0; k < 4; k++) a = fmaf(w1s[j * 4 + k], qn[k], a);
        h[j] = fmaxf(a, 0.f);
    }
    for (int nO = 0; nO < 4; nO++) {
        float a = b2s[nO];
        for (int j = 0; j < 32; j++) a = fmaf(w2s[nO * 32 + j], h[j], a);
        out[(size_t)b * 4 + nO] = a;
    }
}

// ---------------------------------------------------------------------------
extern "C" void kernel_launch(void* const* d_in, const int* in_sizes, int n_in,
                              void* d_out, int out_size, void* d_ws, size_t ws_size,
                              hipStream_t stream) {
    const float* x     = (const float*)d_in[0];
    const float* w1    = (const float*)d_in[1];
    const float* bn1g  = (const float*)d_in[2];
    const float* bn1b  = (const float*)d_in[3];
    const float* w2    = (const float*)d_in[4];
    const float* bn2g  = (const float*)d_in[5];
    const float* bn2b  = (const float*)d_in[6];
    const float* mapw  = (const float*)d_in[7];
    const float* mapb  = (const float*)d_in[8];
    const float* qw    = (const float*)d_in[9];
    const float* bnqg  = (const float*)d_in[10];
    const float* bnqb  = (const float*)d_in[11];
    const float* fc1w  = (const float*)d_in[12];
    const float* fc1b  = (const float*)d_in[13];
    const float* fc2w  = (const float*)d_in[14];
    const float* fc2b  = (const float*)d_in[15];
    float* out = (float*)d_out;

    // Workspace layout (floats)
    float* stats = (float*)d_ws;                 // 64
    float* xmap  = stats + 64;                   // B*4
    float* qv    = xmap + (size_t)BATCH * 4;     // B*4
    float* a1    = qv   + (size_t)BATCH * 4;     // B*8*14*14
    float* a2    = a1   + (size_t)BATCH * 8 * 14 * 14;  // B*784

    hipMemsetAsync(stats, 0, 64 * sizeof(float), stream);

    k1_conv1_stats<<<dim3(BATCH * 784 / 256), dim3(256), 0, stream>>>(x, w1, stats);
    k2_conv1_bn_pool<<<dim3(BATCH * 196 / 256), dim3(256), 0, stream>>>(x, w1, bn1g, bn1b, stats, a1);
    k3_conv2_stats<<<dim3(BATCH * 196 / 256), dim3(256), 0, stream>>>(a1, w2, stats);
    k4_conv2_bn_pool<<<dim3(BATCH * 49 / 256), dim3(256), 0, stream>>>(a1, w2, bn2g, bn2b, stats, a2);
    k5_map_wmma<<<dim3(BATCH / 64), dim3(128), 0, stream>>>(a2, mapw, mapb, xmap);
    k6_quantum<<<dim3(BATCH / 256), dim3(256), 0, stream>>>(xmap, qw, qv, stats);
    k7_head<<<dim3(BATCH / 256), dim3(256), 0, stream>>>(qv, stats, bnqg, bnqb,
                                                         fc1w, fc1b, fc2w, fc2b, out);
}
